// MultiheadAttention_7876970021021
// MI455X (gfx1250) — matrix operand order repaired and checked by reading, not compiled, not run on hardware
//
#include <hip/hip_runtime.h>
#include <hip/hip_bf16.h>

// ---------------------------------------------------------------------------
// MI455X (gfx1250) fused multi-head attention, wave32 + WMMA f16 (f32 acc),
// with ASYNC global->LDS staging of K/V tiles (ASYNCcnt path).
//
// Sizes: B=4, S=2048, HIDDEN=1024, HEADS=16, HEAD=64.
// d_out = [ctx (B*S*HIDDEN f32)] ++ [attn (B*H*S*S f32)]
// d_ws  = Qh f16 [bh][s][64] (16MB) ++ Kh f16 (16MB) ++ Vt f16 [bh][64][s] (16MB)
// ---------------------------------------------------------------------------

#define HIDDEN 1024
#define HEADS  16
#define HEAD   64
#define BATCH  4
#define SEQ    2048
#define AW     4            // waves per block in attention kernel

typedef _Float16 half16v __attribute__((ext_vector_type(16)));
typedef _Float16 half8v  __attribute__((ext_vector_type(8)));
typedef float    float8v __attribute__((ext_vector_type(8)));
typedef float    float4v __attribute__((ext_vector_type(4)));

static __device__ __forceinline__ float8v wmma_f16(const half16v& a, const half16v& b,
                                                   const float8v& c) {
    return __builtin_amdgcn_wmma_f32_16x16x32_f16(false, a, false, b, (short)0, c,
                                                  false, false);
}

// Async per-lane 16B copy global -> LDS (tracked by ASYNCcnt).
// ldsoff = LDS byte offset (low 32 bits of the generic shared pointer).
static __device__ __forceinline__ void async_copy16(unsigned ldsoff,
                                                    const _Float16* gsrc) {
    asm volatile("global_load_async_to_lds_b128 %0, %1, off"
                 :: "v"(ldsoff), "v"(gsrc)
                 : "memory");
}
static __device__ __forceinline__ void async_wait0() {
    asm volatile("s_wait_asynccnt 0" ::: "memory");
}

// ===========================================================================
// Kernel 1: QKV projection.  C[m,n] = sum_k X[m,k]*W[n,k] + bias[n]
// Block = 128 threads (4 waves).  Block tile 64(M) x 64(N); each wave 16x64.
// A-frag (16x32 f16): lane m = l&15, kb = (l<16)?0:8; elems 0-7 = K kb..kb+7,
//                     elems 8-15 = K kb+16..kb+23  (contiguous 16B chunks).
// B-frag (32x16 f16): lane n = l&15, kb = (l<16)?0:8; same K order as A-frag.
// ===========================================================================
__global__ __launch_bounds__(128)
void qkv_proj(const float* __restrict__ qin, const float* __restrict__ kin,
              const float* __restrict__ vin,
              const float* __restrict__ wq, const float* __restrict__ bq,
              const float* __restrict__ wk, const float* __restrict__ bk,
              const float* __restrict__ wv, const float* __restrict__ bv,
              _Float16* __restrict__ qh, _Float16* __restrict__ kh,
              _Float16* __restrict__ vt) {
    const int lane = threadIdx.x & 31;
    const int wid  = threadIdx.x >> 5;
    const int l15  = lane & 15;
    const int hi   = lane >> 4;

    const int sel = blockIdx.z;
    const float* X;  const float* W;  const float* Bv;
    if (sel == 0)      { X = qin; W = wq; Bv = bq; }
    else if (sel == 1) { X = kin; W = wk; Bv = bk; }
    else               { X = vin; W = wv; Bv = bv; }

    const int m0 = blockIdx.x * 64 + wid * 16;   // this wave's 16 rows
    const int n0 = blockIdx.y * 64;

    const int kbA = hi ? 8 : 0;
    const int kbB = hi ? 8 : 0;

    const float* xrow = X + (size_t)(m0 + l15) * HIDDEN;

    float8v acc[4] = {};
    for (int kc = 0; kc < HIDDEN; kc += 32) {
        __builtin_prefetch(xrow + kc + 64, 0, 0);   // global_prefetch_b8
        // A fragment: X rows, f32 -> f16
        float4v a0 = *(const float4v*)(xrow + kc + kbA);
        float4v a1 = *(const float4v*)(xrow + kc + kbA + 4);
        float4v a2 = *(const float4v*)(xrow + kc + kbA + 16);
        float4v a3 = *(const float4v*)(xrow + kc + kbA + 20);
        half16v A;
#pragma unroll
        for (int j = 0; j < 4; ++j) {
            A[j]      = (_Float16)a0[j];
            A[4 + j]  = (_Float16)a1[j];
            A[8 + j]  = (_Float16)a2[j];
            A[12 + j] = (_Float16)a3[j];
        }
#pragma unroll
        for (int nt = 0; nt < 4; ++nt) {
            const float* wrow =
                W + (size_t)(n0 + nt * 16 + l15) * HIDDEN + kc + kbB;
            float4v w0 = *(const float4v*)(wrow);
            float4v w1 = *(const float4v*)(wrow + 4);
            float4v w2 = *(const float4v*)(wrow + 16);
            float4v w3 = *(const float4v*)(wrow + 20);
            half16v Bf;
#pragma unroll
            for (int j = 0; j < 4; ++j) {
                Bf[j]      = (_Float16)w0[j];
                Bf[4 + j]  = (_Float16)w1[j];
                Bf[8 + j]  = (_Float16)w2[j];
                Bf[12 + j] = (_Float16)w3[j];
            }
            acc[nt] = wmma_f16(A, Bf, acc[nt]);
        }
    }

    // Epilogue: +bias, scatter into head-major f16 layouts.
    // D layout: reg r, lanes 0-15 -> row r, lanes 16-31 -> row r+8; col = l&15.
#pragma unroll
    for (int nt = 0; nt < 4; ++nt) {
        const int n  = n0 + nt * 16 + l15;
        const float bb = Bv[n];
        const int h = n >> 6, d = n & 63;
        if (sel < 2) {
            _Float16* dst = (sel == 0) ? qh : kh;
#pragma unroll
            for (int r = 0; r < 8; ++r) {
                const int m = m0 + r + 8 * hi;
                const int b = m >> 11, s = m & 2047;
                dst[((size_t)(b * HEADS + h) * SEQ + s) * HEAD + d] =
                    (_Float16)(acc[nt][r] + bb);
            }
        } else {
            // V transposed: vt[bh][d][s]; this lane owns 8 consecutive s.
            const int m = m0 + 8 * hi;
            const int b = m >> 11, s = m & 2047;
            half8v pk;
#pragma unroll
            for (int r = 0; r < 8; ++r) pk[r] = (_Float16)(acc[nt][r] + bb);
            *(half8v*)(vt + ((size_t)(b * HEADS + h) * HEAD + d) * SEQ + s) = pk;
        }
    }
}

// ===========================================================================
// Kernel 2: fused attention.  Block = 4 waves, all on the same (b,h); each
// wave owns 16 query rows.  K/V 32-key chunks are staged into LDS ONCE per
// block via GLOBAL_LOAD_ASYNC_TO_LDS_B128 (4x less L2 read traffic), fenced
// with s_wait_asynccnt + barrier.  WMMA operands are gathered from LDS.
// Pass 1: online softmax stats (m,l).  Pass 2: recompute scores (L2/LDS-hot),
// write normalized probs to attn, relayout P via LDS, accumulate P*V.
// ===========================================================================
__device__ __forceinline__ float8v score_tile(const half16v qa[2],
                                              const _Float16* krow) {
    float8v acc = {};
#pragma unroll
    for (int dc = 0; dc < 2; ++dc) {
        half8v x0 = *(const half8v*)(krow + dc * 32);
        half8v x1 = *(const half8v*)(krow + dc * 32 + 16);
        half16v Bf;
#pragma unroll
        for (int j = 0; j < 8; ++j) { Bf[j] = x0[j]; Bf[8 + j] = x1[j]; }
        acc = wmma_f16(qa[dc], Bf, acc);
    }
    return acc;
}

__global__ __launch_bounds__(128)
void attn_fused(const _Float16* __restrict__ qh, const _Float16* __restrict__ kh,
                const _Float16* __restrict__ vt, const int* __restrict__ mask,
                float* __restrict__ attn, float* __restrict__ ctx) {
    // Padded rows keep 16B alignment for b128 DS ops and spread banks.
    __shared__ __align__(16) _Float16 ktile[32][72];      // 32 keys x 64 d
    __shared__ __align__(16) _Float16 vtile[64][40];      // 64 d x 32 keys
    __shared__ __align__(16) _Float16 ptile[AW][16][40];  // per-wave P relayout

    const int tid  = threadIdx.x;
    const int lane = tid & 31;
    const int wid  = tid >> 5;
    const int l15  = lane & 15;
    const int hi   = lane >> 4;

    const int h  = blockIdx.y;
    const int b  = blockIdx.z;
    const int bh = b * HEADS + h;
    const int q0 = (blockIdx.x * AW + wid) * 16;
    const float scale = 0.125f;   // 1/sqrt(64)

    const int kbA = hi ? 8 : 0;   // A-fragment K split
    const int kbB = hi ? 8 : 0;   // B-fragment K split

    // LDS byte offsets (low 32 bits of the flat shared pointer == LDS offset).
    const unsigned kbase = (unsigned)(unsigned long long)&ktile[0][0];
    const unsigned vbase = (unsigned)(unsigned long long)&vtile[0][0];

    // Async-staging index split: 256 x 16B segments per tile, 2 per thread.
    const int seg0 = tid * 2;

    // Cache this wave's Q fragments for both 32-wide d chunks (d = 0..63).
    const _Float16* qrow = qh + ((size_t)bh * SEQ + q0 + l15) * HEAD;
    half16v qa[2];
#pragma unroll
    for (int dc = 0; dc < 2; ++dc) {
        half8v c0 = *(const half8v*)(qrow + dc * 32 + kbA);
        half8v c1 = *(const half8v*)(qrow + dc * 32 + kbA + 16);
#pragma unroll
        for (int j = 0; j < 8; ++j) { qa[dc][j] = c0[j]; qa[dc][8 + j] = c1[j]; }
    }

    const int* mrow = mask + (size_t)b * SEQ * SEQ;
    const _Float16* kbh = kh + (size_t)bh * SEQ * HEAD;
    const _Float16* vbh = vt + (size_t)bh * HEAD * SEQ;

    float m[8], l[8];
#pragma unroll
    for (int r = 0; r < 8; ++r) { m[r] = -3.0e38f; l[r] = 0.0f; }

    // ---------------- Pass 1: online max / sum ----------------
    for (int k0 = 0; k0 < SEQ; k0 += 32) {
        __syncthreads();   // previous chunk fully consumed
        // Stage K chunk [32 keys][64 d] f16: seg = key*8 + s8 (16B each).
#pragma unroll
        for (int i = 0; i < 2; ++i) {
            const int seg = seg0 + i;
            const int key = seg >> 3, s8 = seg & 7;
            async_copy16(kbase + (unsigned)(key * 72 + s8 * 8) * 2,
                         kbh + (size_t)(k0 + key) * HEAD + s8 * 8);
        }
        async_wait0();
        __syncthreads();

        float8v s0 = score_tile(qa, &ktile[l15][kbB]);
        float8v s1 = score_tile(qa, &ktile[16 + l15][kbB]);
#pragma unroll
        for (int r = 0; r < 8; ++r) {
            const int qr = q0 + r + 8 * hi;
            const int* mq = mrow + (size_t)qr * SEQ + k0;
            float v0 = mq[l15]      ? s0[r] * scale : -1.0e9f;
            float v1 = mq[16 + l15] ? s1[r] * scale : -1.0e9f;
            float t = fmaxf(v0, v1);
#pragma unroll
            for (int off = 1; off < 16; off <<= 1)
                t = fmaxf(t, __shfl_xor(t, off, 32));
            const float mn = fmaxf(m[r], t);
            float e = __expf(v0 - mn) + __expf(v1 - mn);
#pragma unroll
            for (int off = 1; off < 16; off <<= 1)
                e += __shfl_xor(e, off, 32);
            l[r] = l[r] * __expf(m[r] - mn) + e;
            m[r] = mn;
        }
    }
    float rl[8];
#pragma unroll
    for (int r = 0; r < 8; ++r) rl[r] = 1.0f / l[r];

    // ---------------- Pass 2: probs out + P*V ----------------
    float8v cacc[4] = {};
    for (int k0 = 0; k0 < SEQ; k0 += 32) {
        __syncthreads();
        // Stage K chunk and V chunk [64 d][32 keys] f16 (seg = d*4 + s4).
#pragma unroll
        for (int i = 0; i < 2; ++i) {
            const int seg = seg0 + i;
            const int key = seg >> 3, s8 = seg & 7;
            async_copy16(kbase + (unsigned)(key * 72 + s8 * 8) * 2,
                         kbh + (size_t)(k0 + key) * HEAD + s8 * 8);
            const int d = seg >> 2, s4 = seg & 3;
            async_copy16(vbase + (unsigned)(d * 40 + s4 * 8) * 2,
                         vbh + (size_t)d * SEQ + k0 + s4 * 8);
        }
        async_wait0();
        __syncthreads();

#pragma unroll
        for (int kt = 0; kt < 2; ++kt) {
            float8v sc = score_tile(qa, &ktile[kt * 16 + l15][kbB]);
#pragma unroll
            for (int r = 0; r < 8; ++r) {
                const int qr = q0 + r + 8 * hi;
                const int kc = k0 + kt * 16 + l15;
                const int mk = mrow[(size_t)qr * SEQ + kc];
                const float sv = mk ? sc[r] * scale : -1.0e9f;
                const float p = __expf(sv - m[r]) * rl[r];
                attn[((size_t)bh * SEQ + qr) * SEQ + kc] = p;      // coalesced 64B
                ptile[wid][r + 8 * hi][kt * 16 + l15] = (_Float16)p;
            }
        }
        // Re-gather P [16 x 32] from LDS in A-fragment layout (per-wave tile,
        // DS ops are in-order within a wave).
        half8v p0 = *(const half8v*)&ptile[wid][l15][kbA];
        half8v p1 = *(const half8v*)&ptile[wid][l15][kbA + 16];
        half16v Pa;
#pragma unroll
        for (int j = 0; j < 8; ++j) { Pa[j] = p0[j]; Pa[8 + j] = p1[j]; }
#pragma unroll
        for (int dt = 0; dt < 4; ++dt) {
            const _Float16* vrow = &vtile[dt * 16 + l15][kbB];
            half8v v0 = *(const half8v*)(vrow);
            half8v v1 = *(const half8v*)(vrow + 16);
            half16v Vb;
#pragma unroll
            for (int j = 0; j < 8; ++j) { Vb[j] = v0[j]; Vb[8 + j] = v1[j]; }
            cacc[dt] = wmma_f16(Pa, Vb, cacc[dt]);
        }
    }

    // ctx out: [B, S, H*D] f32 (probs already normalized; no final divide).
#pragma unroll
    for (int dt = 0; dt < 4; ++dt) {
#pragma unroll
        for (int r = 0; r < 8; ++r) {
            const int qr = q0 + r + 8 * hi;
            const int d  = dt * 16 + l15;
            ctx[((size_t)b * SEQ + qr) * HIDDEN + h * HEAD + d] = cacc[dt][r];
        }
    }
}

// ===========================================================================
extern "C" void kernel_launch(void* const* d_in, const int* in_sizes, int n_in,
                              void* d_out, int out_size, void* d_ws, size_t ws_size,
                              hipStream_t stream) {
    const float* qin  = (const float*)d_in[0];
    const float* kin  = (const float*)d_in[1];
    const float* vin  = (const float*)d_in[2];
    const int*   mask = (const int*)d_in[3];
    const float* wq   = (const float*)d_in[4];
    const float* bq   = (const float*)d_in[5];
    const float* wk   = (const float*)d_in[6];
    const float* bk   = (const float*)d_in[7];
    const float* wv   = (const float*)d_in[8];
    const float* bv   = (const float*)d_in[9];

    float* ctx  = (float*)d_out;
    float* attn = ctx + (size_t)BATCH * SEQ * HIDDEN;

    const size_t elems = (size_t)BATCH * HEADS * SEQ * HEAD;   // 8.4M f16 each
    _Float16* qh = (_Float16*)d_ws;
    _Float16* kh = qh + elems;
    _Float16* vt = kh + elems;

    // QKV projection: grid (M/64, N/64, 3), 128 threads (4 waves).
    dim3 pg((BATCH * SEQ) / 64, HIDDEN / 64, 3);
    qkv_proj<<<pg, dim3(128), 0, stream>>>(qin, kin, vin, wq, bq, wk, bk, wv, bv,
                                           qh, kh, vt);

    // Fused attention: grid (S/(16*AW), H, B), 128 threads (AW waves).
    dim3 ag(SEQ / (16 * AW), HEADS, BATCH);
    attn_fused<<<ag, dim3(128), 0, stream>>>(qh, kh, vt, mask, attn, ctx);
}
